// WindowObjectDetector_22041772163087
// MI455X (gfx1250) — compile-verified, matrix operand order checked by reading, not run on hardware
//
#include <hip/hip_runtime.h>

typedef _Float16 v16h __attribute__((ext_vector_type(16)));
typedef float    v8f  __attribute__((ext_vector_type(8)));
typedef float    v4f  __attribute__((ext_vector_type(4)));

#define IMG_W  1600
#define NCH    3
#define NR     97
#define NCOL   97
#define NP     (NR * NCOL)        // 9409 patches
#define KDIM   12288              // 64*64*3
#define NKB    384                // K blocks of 32
#define NTILES ((NP + 15) / 16)   // 589 M-tiles
#define NCLS   10
#define THRV   0.7f

// ---------------------------------------------------------------------------
// Pack W (12288 x 10, f32) into B-fragment lane order for wmma_f32_16x16x32_f16.
// For lane l: n = l&15, base8 = (l>>4)*8.
// Element t<8  -> k_local = base8 + t
// Element t>=8 -> k_local = base8 + 8 + t   (== base8+16 .. base8+23)
// Layout: pw[kb*512 + l*16 + t], f16, n>=10 zero-padded.
// ---------------------------------------------------------------------------
__global__ __launch_bounds__(256) void pack_w_kernel(const float* __restrict__ W,
                                                     _Float16* __restrict__ pw) {
    int idx = blockIdx.x * blockDim.x + threadIdx.x;
    if (idx >= NKB * 512) return;
    int kb  = idx >> 9;
    int rem = idx & 511;
    int l   = rem >> 4;
    int t   = rem & 15;
    int n     = l & 15;
    int base8 = (l >> 4) * 8;
    int klocal = (t < 8) ? (base8 + t) : (base8 + 8 + t);
    int kg = kb * 32 + klocal;
    float v = (n < NCLS) ? W[(size_t)kg * NCLS + n] : 0.0f;
    pw[idx] = (_Float16)v;
}

// ---------------------------------------------------------------------------
// GEMM: one workgroup (8 waves) per 16-patch tile. Wave w covers K blocks
// [w*48, w*48+48) == pixel rows [w*8, w*8+8) of the patch (6 blocks per row).
// Explicit 2-stage software pipeline: loads for iteration it+1 are issued
// before the convert+WMMA of iteration it, so no WMMA waits on a load issued
// in its own iteration. All offsets are compile-time immediates.
// Per-wave v8f accumulators reduced across waves in LDS.
// ---------------------------------------------------------------------------
__global__ __launch_bounds__(256) void gemm_kernel(const float* __restrict__ img,
                                                   const _Float16* __restrict__ pw,
                                                   const float* __restrict__ bias,
                                                   float* __restrict__ logits) {
    __shared__ float red[8 * 256];

    const int tid   = threadIdx.x;
    const int wave  = tid >> 5;
    const int lane  = tid & 31;
    const int m     = lane & 15;
    const int base8 = (lane >> 4) * 8;

    const int p0 = blockIdx.x * 16;
    int p = p0 + m;
    if (p > NP - 1) p = NP - 1;           // clamp tail rows (store is guarded)
    const int pr = p / NCOL;
    const int pc = p - pr * NCOL;

    // Per-lane base pointers; all subsequent accesses are constant offsets.
    const float* __restrict__ abase =
        img + ((size_t)(pr * 16 + wave * 8) * IMG_W + pc * 16) * NCH + base8;
    const _Float16* __restrict__ bbase =
        pw + (size_t)wave * 48 * 512 + (size_t)lane * 16;

    v8f acc = {};

    // double-buffered raw A chunks and B fragments
    v4f  ar[2][4];
    v16h bf[2];

    // prologue: stage iteration 0
    ar[0][0] = *(const v4f*)(abase);
    ar[0][1] = *(const v4f*)(abase + 4);
    ar[0][2] = *(const v4f*)(abase + 16);
    ar[0][3] = *(const v4f*)(abase + 20);
    bf[0]    = *(const v16h*)(bbase);

#pragma unroll
    for (int it = 0; it < 48; ++it) {
        const int cur = it & 1;
        const int nxt = cur ^ 1;

        // issue loads for iteration it+1 first (hidden behind cvt + WMMA of it)
        if (it < 47) {
            const int it1 = it + 1;
            const int il  = it1 / 6;
            const int jb  = it1 - il * 6;
            const float* a0 = abase + il * (IMG_W * NCH) + jb * 32;
            ar[nxt][0] = *(const v4f*)(a0);
            ar[nxt][1] = *(const v4f*)(a0 + 4);
            ar[nxt][2] = *(const v4f*)(a0 + 16);
            ar[nxt][3] = *(const v4f*)(a0 + 20);
            bf[nxt]    = *(const v16h*)(bbase + (size_t)it1 * 512);
        }

        // convert the staged raw A chunks of iteration it to f16 fragment order
        v16h a;
#pragma unroll
        for (int t = 0; t < 4; ++t) {
            a[t]      = (_Float16)ar[cur][0][t];
            a[4 + t]  = (_Float16)ar[cur][1][t];
            a[8 + t]  = (_Float16)ar[cur][2][t];
            a[12 + t] = (_Float16)ar[cur][3][t];
        }

        acc = __builtin_amdgcn_wmma_f32_16x16x32_f16(
            false, a, false, bf[cur], (short)0, acc, false, false);
    }

    // C/D layout: lanes 0-15 vgpr v -> (M=v, N=lane); lanes 16-31 -> (M=v+8, N=lane-16)
#pragma unroll
    for (int v = 0; v < 8; ++v) {
        const int row = v + base8;
        red[wave * 256 + row * 16 + m] = acc[v];
    }
    __syncthreads();

    {
        const int e = tid;                 // 256 elements of the 16x16 tile
        float s = 0.0f;
#pragma unroll
        for (int w = 0; w < 8; ++w) s += red[w * 256 + e];
        const int row = e >> 4;
        const int col = e & 15;
        if (p0 + row < NP) {
            float bb = (col < NCLS) ? bias[col] : 0.0f;
            logits[(size_t)(p0 + row) * 16 + col] = s + bb;
        }
    }
}

// ---------------------------------------------------------------------------
// bbox init: (10 classes x 4) -> {INT_MAX, INT_MAX, -1, -1}
// ---------------------------------------------------------------------------
__global__ void init_bbox_kernel(int* __restrict__ bbox) {
    int i = threadIdx.x;
    if (i < NCLS * 4) bbox[i] = ((i & 3) < 2) ? 0x7fffffff : -1;
}

// ---------------------------------------------------------------------------
// Per-patch softmax over 10 classes, prob -> d_out, threshold -> bbox atomics.
// ---------------------------------------------------------------------------
__global__ __launch_bounds__(256) void softmax_bbox_kernel(const float* __restrict__ logits,
                                                           float* __restrict__ prob_out,
                                                           int* __restrict__ bbox) {
    int p = blockIdx.x * blockDim.x + threadIdx.x;
    if (p >= NP) return;

    float l[NCLS];
    float mx = -3.4e38f;
#pragma unroll
    for (int n = 0; n < NCLS; ++n) {
        l[n] = logits[(size_t)p * 16 + n];
        mx = fmaxf(mx, l[n]);
    }
    float s = 0.0f;
#pragma unroll
    for (int n = 0; n < NCLS; ++n) {
        l[n] = __expf(l[n] - mx);
        s += l[n];
    }
    const float inv = 1.0f / s;
    const int r = p / NCOL;
    const int c = p - r * NCOL;
#pragma unroll
    for (int n = 0; n < NCLS; ++n) {
        float pr = l[n] * inv;
        prob_out[(size_t)p * NCLS + n] = pr;
        if (pr > THRV) {
            atomicMin(&bbox[n * 4 + 0], r);
            atomicMin(&bbox[n * 4 + 1], c);
            atomicMax(&bbox[n * 4 + 2], r);
            atomicMax(&bbox[n * 4 + 3], c);
        }
    }
}

// ---------------------------------------------------------------------------
// bbox ints -> float tail of d_out
// ---------------------------------------------------------------------------
__global__ void finalize_bbox_kernel(const int* __restrict__ bbox,
                                     float* __restrict__ out) {
    int i = threadIdx.x;
    if (i < NCLS * 4) out[i] = (float)bbox[i];
}

extern "C" void kernel_launch(void* const* d_in, const int* in_sizes, int n_in,
                              void* d_out, int out_size, void* d_ws, size_t ws_size,
                              hipStream_t stream) {
    const float* x    = (const float*)d_in[0];  // (1,1600,1600,3) f32
    const float* W    = (const float*)d_in[1];  // (12288,10) f32
    const float* bias = (const float*)d_in[2];  // (10,) f32
    float* out = (float*)d_out;                 // 9409*10 prob + 40 bbox

    // workspace layout
    char* ws = (char*)d_ws;
    _Float16* pw   = (_Float16*)ws;                           // 384*512 f16 = 393216 B
    float*    lgt  = (float*)(ws + 393216);                   // 9424*16 f32 = 603136 B
    int*      bbox = (int*)(ws + 393216 + 603136);            // 40 int32    = 160 B

    pack_w_kernel<<<(NKB * 512 + 255) / 256, 256, 0, stream>>>(W, pw);
    init_bbox_kernel<<<1, 64, 0, stream>>>(bbox);
    gemm_kernel<<<NTILES, 256, 0, stream>>>(x, pw, bias, lgt);
    softmax_bbox_kernel<<<(NP + 255) / 256, 256, 0, stream>>>(lgt, out, bbox);
    finalize_bbox_kernel<<<1, 64, 0, stream>>>(bbox, out + (size_t)NP * NCLS);
}
